// Encoder_5686536700540
// MI455X (gfx1250) — matrix operand order repaired and checked by reading, not compiled, not run on hardware
//
#include <hip/hip_runtime.h>
#include <cstdint>
#include <cstddef>

// ---------------- problem constants ----------------
#define B_  4
#define S_  1024
#define D_  768
#define H_  12
#define HD_ 64
#define HID_ 768
#define DEPTH_ 6
#define M_  (B_ * S_)          // 4096 rows
#define MD_ ((size_t)M_ * D_)  // 3145728 elements
#define WSZ_ ((size_t)D_ * HID_) // 589824 weight elements
#define EPS_ 1e-5f
#define SCALE_ 0.125f          // 64^-0.5

typedef __attribute__((ext_vector_type(16))) __bf16 v16bf;
typedef __attribute__((ext_vector_type(8)))  float  v8f;
typedef __attribute__((ext_vector_type(4)))  unsigned int u32x4;
typedef __attribute__((ext_vector_type(8)))  int  i32x8;
typedef __attribute__((ext_vector_type(4)))  int  i32x4;

union Frag16 { uint32_t u[8]; v16bf v; };

#define WMMA_BF16(a, b, c) \
  __builtin_amdgcn_wmma_f32_16x16x32_bf16(false, (a), false, (b), (short)0, (c), false, false)

#define HAVE_TDM_ (__has_builtin(__builtin_amdgcn_tensor_load_to_lds))

static __device__ __forceinline__ v8f vzero8() {
  v8f z;
#pragma unroll
  for (int i = 0; i < 8; ++i) z[i] = 0.0f;
  return z;
}

#if HAVE_TDM_
// Issue one 2-D TDM tile load: tile_d0 x tile_d1 elements of 2 bytes,
// row stride (in elements) = stride0, from gaddr into LDS offset lds_off.
// D# bit layout per CDNA5 ISA 8.3/8.4 (group0: count/lds/global/type=2,
// group1: data_size=1(2B), dims/strides). Trailing groups zero (2-D tensor).
// This toolchain exposes the 6-arg builtin form:
//   (u32x4 g0, i32x8 g1, i32x4, i32x4, i32x8, i32 cpol)
static __device__ __forceinline__ void tdm_load_2d_bf16(
    unsigned lds_off, const void* gaddr, unsigned tensor_d0, unsigned tensor_d1,
    unsigned tile_d0, unsigned tile_d1, unsigned stride0) {
  uint64_t ga = (uint64_t)gaddr;
  u32x4 g0;
  g0[0] = 1u;                                   // count=1, user mode
  g0[1] = lds_off;                              // lds_addr (bytes)
  g0[2] = (unsigned)(ga & 0xFFFFFFFFu);         // global_addr[31:0]
  g0[3] = (unsigned)((ga >> 32) & 0x01FFFFFFu)  // global_addr[56:32]
          | (2u << 30);                         // type = 2 ("image")
  i32x8 g1;
  g1[0] = (int)(1u << 16);                      // data_size=1 (2 bytes)
  g1[1] = (int)((tensor_d0 & 0xFFFFu) << 16);   // tensor_dim0[15:0] @48
  g1[2] = (int)((tensor_d0 >> 16) | ((tensor_d1 & 0xFFFFu) << 16));
  g1[3] = (int)((tensor_d1 >> 16) | (tile_d0 << 16));
  g1[4] = (int)(tile_d1 & 0xFFFFu);             // tile_dim1 (tile_dim2=0)
  g1[5] = (int)stride0;                         // tensor_dim0_stride[31:0]
  g1[6] = 0;                                    // stride0[47:32] | stride1[15:0]
  g1[7] = 0;                                    // stride1[47:16]
  i32x4 z4; z4[0] = 0; z4[1] = 0; z4[2] = 0; z4[3] = 0;
  i32x8 z8;
#pragma unroll
  for (int i = 0; i < 8; ++i) z8[i] = 0;
  __builtin_amdgcn_tensor_load_to_lds(g0, g1, z4, z4, z8, 0);
}
#endif

// ---------------- embedding + positional ----------------
__global__ __launch_bounds__(256) void embed_kernel(
    const int* __restrict__ x, const float* __restrict__ emb,
    const float* __restrict__ pos, float* __restrict__ h) {
  size_t e = (size_t)blockIdx.x * 256 + threadIdx.x;
  if (e >= MD_) return;
  int bs = (int)(e / D_);
  int d  = (int)(e % D_);
  int s  = bs % S_;
  int tok = x[bs];
  h[e] = emb[(size_t)tok * D_ + d] + pos[(size_t)s * D_ + d];
}

// ---------------- fp32 -> bf16 weight convert ----------------
__global__ __launch_bounds__(256) void conv_kernel(
    const float* __restrict__ in, __bf16* __restrict__ out) {
  size_t e = (size_t)blockIdx.x * 256 + threadIdx.x;
  if (e >= WSZ_) return;
  out[e] = (__bf16)in[e];
}

// ---------------- LayerNorm over D=768 (one block per row) ----------------
__global__ __launch_bounds__(256) void ln_kernel(
    const float* __restrict__ x, const float* __restrict__ g,
    const float* __restrict__ bta, float* __restrict__ of32,
    __bf16* __restrict__ obf) {
  __shared__ float red[256];
  const int row = blockIdx.x, t = threadIdx.x;
  const float* xr = x + (size_t)row * D_;
  float v0 = xr[t], v1 = xr[t + 256], v2 = xr[t + 512];
  red[t] = v0 + v1 + v2;
  __syncthreads();
#pragma unroll
  for (int o = 128; o > 0; o >>= 1) {
    if (t < o) red[t] += red[t + o];
    __syncthreads();
  }
  float mean = red[0] * (1.0f / 768.0f);
  __syncthreads();
  float d0 = v0 - mean, d1 = v1 - mean, d2 = v2 - mean;
  red[t] = d0 * d0 + d1 * d1 + d2 * d2;
  __syncthreads();
#pragma unroll
  for (int o = 128; o > 0; o >>= 1) {
    if (t < o) red[t] += red[t + o];
    __syncthreads();
  }
  float inv = rsqrtf(red[0] * (1.0f / 768.0f) + EPS_);
  float y0 = d0 * inv * g[t]       + bta[t];
  float y1 = d1 * inv * g[t + 256] + bta[t + 256];
  float y2 = d2 * inv * g[t + 512] + bta[t + 512];
  size_t base = (size_t)row * D_ + t;
  if (of32) { of32[base] = y0; of32[base + 256] = y1; of32[base + 512] = y2; }
  if (obf)  { obf[base] = (__bf16)y0; obf[base + 256] = (__bf16)y1; obf[base + 512] = (__bf16)y2; }
}

// ---------------- bf16 WMMA GEMM: C[M,N] = A[M,K] @ W[K,N] + bias ----------------
// A tile staged via TDM (tensor_load_to_lds); W tile transposed-on-store to
// match the WMMA B-fragment layout.
// EPI 0: bf16 out (+bias)
// EPI 1: bf16 out (+bias, exact GELU)
// EPI 2: f32 residual in/out: resf += acc + bias
template <int EPI>
__global__ __launch_bounds__(256) void gemm_wmma(
    const __bf16* __restrict__ A, const __bf16* __restrict__ W,
    const float* __restrict__ bias, float* __restrict__ resf,
    __bf16* __restrict__ outb, int M, int N, int K) {
  __shared__ __align__(16) __bf16 sA[128 * 32];
  __shared__ __align__(16) __bf16 sW[64 * 34];
  const int tid = threadIdx.x;
  const int lane = tid & 31, wid = tid >> 5;
  const int wm = wid & 3, wn = wid >> 2;   // 4 x 2 wave grid -> 128 x 64 block tile
  const int m0 = blockIdx.y * 128;
  const int n0 = blockIdx.x * 64;
  const int lm = lane & 15, khalf = lane >> 4;

  v8f acc[2][2];
#pragma unroll
  for (int i = 0; i < 2; ++i)
#pragma unroll
    for (int j = 0; j < 2; ++j) acc[i][j] = vzero8();

  for (int kt = 0; kt < K; kt += 32) {
    // stage A tile [128 x 32]
#if HAVE_TDM_
    if (wid == 0) {
      tdm_load_2d_bf16((unsigned)(uintptr_t)&sA[0],
                       A + (size_t)m0 * K + kt,
                       (unsigned)K, (unsigned)M, 32u, 128u, (unsigned)K);
      __builtin_amdgcn_s_wait_tensorcnt(0);
    }
#else
#pragma unroll
    for (int i = 0; i < 2; ++i) {
      int idx = tid + i * 256;           // 0..511
      int row = idx >> 2, ch = idx & 3;  // 4 uint4 chunks per row
      uint4 va = *(const uint4*)(A + (size_t)(m0 + row) * K + kt + ch * 8);
      *(uint4*)(&sA[row * 32 + ch * 8]) = va;
    }
#endif
    // stage W tile transposed: sW[n][k], 64 x 32 (+pad 2)
    {
      int k = tid >> 3, ng = tid & 7;
      uint4 vw = *(const uint4*)(W + (size_t)(kt + k) * N + n0 + ng * 8);
      const __bf16* hw = (const __bf16*)&vw;
#pragma unroll
      for (int j = 0; j < 8; ++j) sW[(ng * 8 + j) * 34 + k] = hw[j];
    }
    if (kt + 32 < K) {
      __builtin_prefetch(W + (size_t)(kt + 32 + (tid >> 3)) * N + n0, 0, 1);
    }
    __syncthreads();

    Frag16 a[2], b[2];
#pragma unroll
    for (int si = 0; si < 2; ++si) {
      int row = wm * 32 + si * 16 + lm;
#pragma unroll
      for (int p = 0; p < 8; ++p) {
        int k0 = khalf * 8 + (p < 4 ? 2 * p : 16 + 2 * (p - 4));
        a[si].u[p] = *(const uint32_t*)(&sA[row * 32 + k0]);
      }
    }
#pragma unroll
    for (int sj = 0; sj < 2; ++sj) {
      int n = wn * 32 + sj * 16 + lm;
#pragma unroll
      for (int p = 0; p < 8; ++p) {
        int k0 = khalf * 8 + (p < 4 ? 2 * p : 16 + 2 * (p - 4));
        b[sj].u[p] = *(const uint32_t*)(&sW[n * 34 + k0]);
      }
    }
#pragma unroll
    for (int si = 0; si < 2; ++si)
#pragma unroll
      for (int sj = 0; sj < 2; ++sj)
        acc[si][sj] = WMMA_BF16(a[si].v, b[sj].v, acc[si][sj]);
    __syncthreads();
  }

  // epilogue
#pragma unroll
  for (int si = 0; si < 2; ++si)
#pragma unroll
    for (int sj = 0; sj < 2; ++sj) {
      int n = n0 + wn * 32 + sj * 16 + lm;
      float bv = bias[n];
#pragma unroll
      for (int r = 0; r < 8; ++r) {
        int row = m0 + wm * 32 + si * 16 + r + 8 * khalf;
        float v = acc[si][sj][r] + bv;
        size_t off = (size_t)row * N + n;
        if (EPI == 0) {
          outb[off] = (__bf16)v;
        } else if (EPI == 1) {
          float gl = 0.5f * v * (1.0f + erff(v * 0.70710678f));
          outb[off] = (__bf16)gl;
        } else {
          resf[off] = resf[off] + v;
        }
      }
    }
}

// ---------------- fused flash attention ----------------
// block = 4 waves (128 thr); each wave owns a 16-row q tile; block covers 64 q rows.
// grid = (S/64, B*H). K tile staged with async global->LDS (ASYNCcnt); V tile
// transposed manually; online softmax; output in the reference's bug-faithful
// [B,H,S,HD]->[B,S,HID] flat order.
__global__ __launch_bounds__(128) void attn_flash(
    const __bf16* __restrict__ qb, const __bf16* __restrict__ kb,
    const __bf16* __restrict__ vb, const int* __restrict__ mask,
    __bf16* __restrict__ ob) {
  __shared__ __align__(16) __bf16 sK[32 * 64];
  __shared__ __align__(16) __bf16 sV[64 * 36];   // transposed [dim][key]
  __shared__ __align__(16) __bf16 sP[4 * 16 * 32];
  const int tid = threadIdx.x, lane = tid & 31, wid = tid >> 5;
  const int bh = blockIdx.y;
  const int b = bh / H_, h = bh % H_;
  const int q0 = blockIdx.x * 64 + wid * 16;
  const int lm = lane & 15, khalf = lane >> 4;

  // q A-fragments (16 rows x 64 dims -> two 16x32 frags)
  Frag16 aq[2];
  {
    int row = q0 + lm;
    const __bf16* qp = qb + (size_t)(b * S_ + row) * HID_ + h * HD_;
#pragma unroll
    for (int kk = 0; kk < 2; ++kk)
#pragma unroll
      for (int p = 0; p < 8; ++p) {
        int k0 = kk * 32 + khalf * 8 + (p < 4 ? 2 * p : 16 + 2 * (p - 4));
        aq[kk].u[p] = *(const uint32_t*)(qp + k0);
      }
  }

  float mI[8], lI[8];
  v8f accO[4];
#pragma unroll
  for (int r = 0; r < 8; ++r) { mI[r] = -3.0e38f; lI[r] = 0.0f; }
#pragma unroll
  for (int g = 0; g < 4; ++g) accO[g] = vzero8();

  for (int kb0 = 0; kb0 < S_; kb0 += 32) {
    // stage K tile [key][dim] via async global->LDS copies (b128 per lane),
    // V tile transposed [dim][key] via regular loads + LDS stores.
#pragma unroll
    for (int i = 0; i < 2; ++i) {
      int idx = tid + i * 128;            // 0..255
      int key = idx >> 3, ch = idx & 7;   // 8 uint4 chunks per key
      size_t gbase = (size_t)(b * S_ + kb0 + key) * HID_ + h * HD_ + ch * 8;
      unsigned lds_off = (unsigned)(uintptr_t)&sK[key * 64 + ch * 8];
      uint64_t ga = (uint64_t)(kb + gbase);
      asm volatile("global_load_async_to_lds_b128 %0, %1, off"
                   :: "v"(lds_off), "v"(ga) : "memory");
      uint4 vv = *(const uint4*)(vb + gbase);
      const __bf16* hv = (const __bf16*)&vv;
#pragma unroll
      for (int j = 0; j < 8; ++j) sV[(ch * 8 + j) * 36 + key] = hv[j];
    }
    asm volatile("s_wait_asynccnt 0" ::: "memory");
    __syncthreads();

    // scores: s0 = keys [kb0..kb0+15], s1 = keys [kb0+16..kb0+31]
    v8f s0 = vzero8(), s1 = vzero8();
#pragma unroll
    for (int kk = 0; kk < 2; ++kk) {
      Frag16 bk0, bk1;
#pragma unroll
      for (int p = 0; p < 8; ++p) {
        int d0 = kk * 32 + khalf * 8 + (p < 4 ? 2 * p : 16 + 2 * (p - 4));
        bk0.u[p] = *(const uint32_t*)(&sK[lm * 64 + d0]);
        bk1.u[p] = *(const uint32_t*)(&sK[(16 + lm) * 64 + d0]);
      }
      s0 = WMMA_BF16(aq[kk].v, bk0.v, s0);
      s1 = WMMA_BF16(aq[kk].v, bk1.v, s1);
    }

    // mask BEFORE scale (faithful to reference), then online softmax
    bool pad0 = (mask[b * S_ + kb0 + lm] == 0);
    bool pad1 = (mask[b * S_ + kb0 + 16 + lm] == 0);
    float pf0[8], pf1[8], fac[8];
#pragma unroll
    for (int r = 0; r < 8; ++r) {
      float x0 = (pad0 ? -1.0e20f : s0[r]) * SCALE_;
      float x1 = (pad1 ? -1.0e20f : s1[r]) * SCALE_;
      float mx = fmaxf(x0, x1);
#pragma unroll
      for (int o = 8; o >= 1; o >>= 1) mx = fmaxf(mx, __shfl_xor(mx, o, 32));
      float mnew = fmaxf(mI[r], mx);
      fac[r] = __expf(mI[r] - mnew);
      float p0 = __expf(x0 - mnew), p1 = __expf(x1 - mnew);
      float rs = p0 + p1;
#pragma unroll
      for (int o = 8; o >= 1; o >>= 1) rs += __shfl_xor(rs, o, 32);
      lI[r] = lI[r] * fac[r] + rs;
      mI[r] = mnew;
      pf0[r] = p0; pf1[r] = p1;
    }
#pragma unroll
    for (int g = 0; g < 4; ++g)
#pragma unroll
      for (int r = 0; r < 8; ++r) accO[g][r] *= fac[r];

    // bounce P through per-wave LDS to re-layout into a WMMA A fragment
    __bf16* pw = &sP[wid * 16 * 32];
#pragma unroll
    for (int r = 0; r < 8; ++r) {
      pw[(r + 8 * khalf) * 32 + lm] = (__bf16)pf0[r];
      pw[(r + 8 * khalf) * 32 + 16 + lm] = (__bf16)pf1[r];
    }
    asm volatile("s_wait_dscnt 0" ::: "memory");
    Frag16 ap;
#pragma unroll
    for (int p = 0; p < 8; ++p) {
      int k0 = khalf * 8 + (p < 4 ? 2 * p : 16 + 2 * (p - 4));
      ap.u[p] = *(const uint32_t*)(&pw[lm * 32 + k0]);
    }
    // o += P[16x32] @ V[32x64]  (4 dim-groups of 16)
#pragma unroll
    for (int g = 0; g < 4; ++g) {
      Frag16 bv;
#pragma unroll
      for (int p = 0; p < 8; ++p) {
        int koff = khalf * 8 + (p < 4 ? 2 * p : 16 + 2 * (p - 4));
        bv.u[p] = *(const uint32_t*)(&sV[(g * 16 + lm) * 36 + koff]);
      }
      accO[g] = WMMA_BF16(ap.v, bv.v, accO[g]);
    }
    __syncthreads();
  }

  // normalize and store in the buggy flat order: ((b*H + h)*S + s)*HD + d
#pragma unroll
  for (int g = 0; g < 4; ++g)
#pragma unroll
    for (int r = 0; r < 8; ++r) {
      int srow = q0 + r + 8 * khalf;
      int d = g * 16 + lm;
      float v = accO[g][r] / lI[r];
      ob[((size_t)(b * H_ + h) * S_ + srow) * HD_ + d] = (__bf16)v;
    }
}

// ---------------- driver ----------------
extern "C" void kernel_launch(void* const* d_in, const int* in_sizes, int n_in,
                              void* d_out, int out_size, void* d_ws, size_t ws_size,
                              hipStream_t stream) {
  const int*   x      = (const int*)d_in[0];
  const int*   mask   = (const int*)d_in[1];
  const float* emb    = (const float*)d_in[2];
  const float* pos    = (const float*)d_in[3];
  const float* attn_g = (const float*)d_in[4];
  const float* attn_b = (const float*)d_in[5];
  const float* Wq = (const float*)d_in[6];
  const float* bq = (const float*)d_in[7];
  const float* Wk = (const float*)d_in[8];
  const float* bk = (const float*)d_in[9];
  const float* Wv = (const float*)d_in[10];
  const float* bv = (const float*)d_in[11];
  const float* Wo = (const float*)d_in[12];
  const float* bo = (const float*)d_in[13];
  const float* ff_g = (const float*)d_in[14];
  const float* ff_b = (const float*)d_in[15];
  const float* W1 = (const float*)d_in[16];
  const float* b1 = (const float*)d_in[17];
  const float* W2 = (const float*)d_in[18];
  const float* b2 = (const float*)d_in[19];
  const float* enc_g = (const float*)d_in[20];
  const float* enc_b = (const float*)d_in[21];

  float* h = (float*)d_out;  // persistent hidden state lives in d_out [M, D] fp32

  // workspace: 4 activation bf16 buffers + 1 bf16 weight buffer (~26.3 MB)
  __bf16* nb  = (__bf16*)d_ws;     // LN-out / attention-out (aliased)
  __bf16* qb  = nb  + MD_;         // q / FFN hidden (aliased)
  __bf16* kbf = qb  + MD_;
  __bf16* vbf = kbf + MD_;
  __bf16* wb  = vbf + MD_;         // bf16 weight scratch (768x768)
  __bf16* ob  = nb;                // alias: LN-out consumed before attention writes
  __bf16* f1b = qb;                // alias: q consumed before FFN hidden written

  const dim3 gGrid(HID_ / 64, M_ / 128);   // (12, 32)
  const dim3 aGrid(S_ / 64, B_ * H_);      // (16, 48)
  const int embBlocks = (int)((MD_ + 255) / 256);
  const int cvtBlocks = (int)((WSZ_ + 255) / 256);

  embed_kernel<<<embBlocks, 256, 0, stream>>>(x, emb, pos, h);

  for (int i = 0; i < DEPTH_; ++i) {
    const size_t wOff = (size_t)i * WSZ_;
    const size_t vOff = (size_t)i * D_;

    // --- attention block ---
    ln_kernel<<<M_, 256, 0, stream>>>(h, attn_g + vOff, attn_b + vOff, nullptr, nb);
    conv_kernel<<<cvtBlocks, 256, 0, stream>>>(Wq + wOff, wb);
    gemm_wmma<0><<<gGrid, 256, 0, stream>>>(nb, wb, bq + vOff, nullptr, qb, M_, HID_, D_);
    conv_kernel<<<cvtBlocks, 256, 0, stream>>>(Wk + wOff, wb);
    gemm_wmma<0><<<gGrid, 256, 0, stream>>>(nb, wb, bk + vOff, nullptr, kbf, M_, HID_, D_);
    conv_kernel<<<cvtBlocks, 256, 0, stream>>>(Wv + wOff, wb);
    gemm_wmma<0><<<gGrid, 256, 0, stream>>>(nb, wb, bv + vOff, nullptr, vbf, M_, HID_, D_);

    attn_flash<<<aGrid, 128, 0, stream>>>(qb, kbf, vbf, mask, ob);

    conv_kernel<<<cvtBlocks, 256, 0, stream>>>(Wo + wOff, wb);
    gemm_wmma<2><<<gGrid, 256, 0, stream>>>(ob, wb, bo + vOff, h, nullptr, M_, D_, HID_);
    ln_kernel<<<M_, 256, 0, stream>>>(h, enc_g, enc_b, h, nullptr);

    // --- FFN block ---
    ln_kernel<<<M_, 256, 0, stream>>>(h, ff_g + vOff, ff_b + vOff, nullptr, nb);
    conv_kernel<<<cvtBlocks, 256, 0, stream>>>(W1 + wOff, wb);
    gemm_wmma<1><<<gGrid, 256, 0, stream>>>(nb, wb, b1 + vOff, nullptr, f1b, M_, HID_, D_);
    conv_kernel<<<cvtBlocks, 256, 0, stream>>>(W2 + wOff, wb);
    gemm_wmma<2><<<gGrid, 256, 0, stream>>>(f1b, wb, b2 + vOff, h, nullptr, M_, D_, HID_);
    ln_kernel<<<M_, 256, 0, stream>>>(h, enc_g, enc_b, h, nullptr);
  }
}
